// SenderImitation_63900523430241
// MI455X (gfx1250) — matrix operand order repaired and checked by reading, not compiled, not run on hardware
//
#include <hip/hip_runtime.h>

#define BATCH 1024
#define ATTR  256
#define HID   512
#define VOCAB 8192
#define MSG_LEN 16
#define G4 (4*HID)   // 2048

typedef __attribute__((ext_vector_type(16))) __bf16 v16bf;
typedef __attribute__((ext_vector_type(8)))  __bf16 v8bf;
typedef __attribute__((ext_vector_type(8)))  float  v8f;

// ---------------- utility kernels ----------------

__global__ void f32_to_bf16_k(const float* __restrict__ src, __bf16* __restrict__ dst, int n) {
    int i = blockIdx.x * blockDim.x + threadIdx.x;
    if (i < n) dst[i] = (__bf16)src[i];
}

__global__ void fill_zero_f32_k(float* __restrict__ p, int n) {
    int i = blockIdx.x * blockDim.x + threadIdx.x;
    if (i < n) p[i] = 0.0f;
}

// ---------------- WMMA GEMM: out[M,N] = A[M,K] @ W[N,K]^T (+bias +bias2 +Wih[:,tok] gather) ----------
// One wave computes a 32x64 tile: 2 A fragments x 4 B fragments -> 8 WMMAs per K-step,
// 12 b128 loads per K-step (1.5 loads/WMMA; B strip traffic halved vs 16x64 tiling).
// A fragment (16-bit 16x32): lane half=lane>>4, row=lane&15; elems 0..7 -> K = k0 + half*16 + e,
//                            elems 8..15 -> K = k0 + half*16 + e  (two contiguous 16B loads)
// B fragment (16-bit 32x16): lane col n = nBase + (lane&15); elems e -> K = k0 + half*16 + e
//                            (one contiguous 32B load, since Wᵀ columns are W rows)
// C/D f32 16x16: elem v -> m = 16*tile + half*8 + v, n = lane&15 within tile.
__global__ __launch_bounds__(256)
void wmma_gemm_bf16(const __bf16* __restrict__ A,      // [M,K] row-major bf16
                    const __bf16* __restrict__ W,      // [N,K] row-major bf16
                    const float*  __restrict__ bias,   // [N] or null
                    const float*  __restrict__ bias2,  // [N] or null
                    const float*  __restrict__ Wih,    // [N, VOCAB] f32 or null (one-hot gather)
                    const int*    __restrict__ tokens, // [M] or null
                    float*        __restrict__ out,    // [M,N] f32
                    __bf16*       __restrict__ out_bf, // [M,N] bf16 mirror or null
                    int M, int N, int K)
{
    const int lane  = threadIdx.x & 31;
    const int wave  = (blockIdx.x * blockDim.x + threadIdx.x) >> 5;
    const int nSuper = N >> 6;                  // # of 64-wide strips
    const int mTile  = wave / nSuper;           // 32-row tile index
    const int nBase  = (wave - mTile * nSuper) << 6;
    const int half   = lane >> 4;
    const int lr     = lane & 15;

    const __bf16* aRow0 = A + (size_t)(mTile * 32 + lr) * K + half * 16;
    const __bf16* aRow1 = aRow0 + (size_t)16 * K;

    v8f acc[2][4] = {};

    for (int k0 = 0; k0 < K; k0 += 32) {
        v16bf a[2];
        {
            v8bf lo = *(const v8bf*)(aRow0 + k0);
            v8bf hi = *(const v8bf*)(aRow0 + k0 + 8);
#pragma unroll
            for (int i = 0; i < 8; ++i) { a[0][i] = lo[i]; a[0][i + 8] = hi[i]; }
        }
        {
            v8bf lo = *(const v8bf*)(aRow1 + k0);
            v8bf hi = *(const v8bf*)(aRow1 + k0 + 8);
#pragma unroll
            for (int i = 0; i < 8; ++i) { a[1][i] = lo[i]; a[1][i + 8] = hi[i]; }
        }

#pragma unroll
        for (int j = 0; j < 4; ++j) {
            const __bf16* bp = W + (size_t)(nBase + j * 16 + lr) * K + k0 + half * 16;
            v16bf b = *(const v16bf*)bp;
            acc[0][j] = __builtin_amdgcn_wmma_f32_16x16x32_bf16(
                false, a[0], false, b, (short)0, acc[0][j], false, false);
            acc[1][j] = __builtin_amdgcn_wmma_f32_16x16x32_bf16(
                false, a[1], false, b, (short)0, acc[1][j], false, false);
        }
    }

    int tok[2][8];
    if (tokens) {
#pragma unroll
        for (int i = 0; i < 2; ++i)
#pragma unroll
            for (int v = 0; v < 8; ++v)
                tok[i][v] = tokens[mTile * 32 + i * 16 + half * 8 + v];
    }

#pragma unroll
    for (int j = 0; j < 4; ++j) {
        const int n = nBase + j * 16 + lr;
        float badd = 0.0f;
        if (bias)  badd += bias[n];
        if (bias2) badd += bias2[n];
        const float* wihRow = Wih ? (Wih + (size_t)n * VOCAB) : nullptr;
#pragma unroll
        for (int i = 0; i < 2; ++i) {
#pragma unroll
            for (int v = 0; v < 8; ++v) {
                const int m = mTile * 32 + i * 16 + half * 8 + v;
                float val = acc[i][j][v] + badd;
                if (wihRow) val += wihRow[tok[i][v]];
                const size_t o = (size_t)m * N + n;
                out[o] = val;
                if (out_bf) out_bf[o] = (__bf16)val;
            }
        }
    }
}

// ---------------- log-softmax + greedy argmax (one block per batch row) ----------------
__global__ __launch_bounds__(256)
void logsoftmax_argmax(float* __restrict__ logits,   // [BATCH, VOCAB] slab for step t (in d_out)
                       const int* __restrict__ trg,  // trg_var [BATCH, MSG_LEN]
                       int t,
                       float* __restrict__ msg,      // [BATCH, MSG_LEN] floats in d_out
                       float* __restrict__ picked,   // [MSG_LEN, BATCH] scratch
                       int*   __restrict__ tokens)   // [BATCH] scratch
{
    __shared__ float smax[256];
    __shared__ int   sidx[256];
    __shared__ float ssum[256];
    const int b   = blockIdx.x;
    const int tid = threadIdx.x;
    float* row = logits + (size_t)b * VOCAB;

    // pass 1: max + first-occurrence argmax
    float vmax = -3.402823466e38f; int imax = 0;
    for (int v = tid; v < VOCAB; v += 256) {
        float x = row[v];
        if (x > vmax) { vmax = x; imax = v; }
    }
    smax[tid] = vmax; sidx[tid] = imax;
    __syncthreads();
    for (int s = 128; s > 0; s >>= 1) {
        if (tid < s) {
            float xo = smax[tid + s]; int io = sidx[tid + s];
            if (xo > smax[tid] || (xo == smax[tid] && io < sidx[tid])) {
                smax[tid] = xo; sidx[tid] = io;
            }
        }
        __syncthreads();
    }
    vmax = smax[0]; imax = sidx[0];

    // pass 2: sum of exp (deterministic fixed-order tree)
    float psum = 0.0f;
    for (int v = tid; v < VOCAB; v += 256) psum += __expf(row[v] - vmax);
    ssum[tid] = psum;
    __syncthreads();
    for (int s = 128; s > 0; s >>= 1) { if (tid < s) ssum[tid] += ssum[tid + s]; __syncthreads(); }
    const float logZ = vmax + __logf(ssum[0]);

    // pass 3: rewrite in place to log-probs
    for (int v = tid; v < VOCAB; v += 256) row[v] = row[v] - logZ;
    __syncthreads();

    if (tid == 0) {
        tokens[b] = imax;
        msg[b * MSG_LEN + t] = (float)imax;
        picked[t * BATCH + b] = row[trg[b * MSG_LEN + t]];
    }
}

// ---------------- LSTM pointwise update ----------------
__global__ __launch_bounds__(256)
void lstm_update(const float* __restrict__ gates, // [B, 4H] order i,f,g,o
                 float* __restrict__ c,
                 float* __restrict__ h,
                 __bf16* __restrict__ hb)
{
    const int idx = blockIdx.x * blockDim.x + threadIdx.x; // B*HID
    const int b = idx >> 9;        // / HID
    const int j = idx & (HID - 1);
    const float* g = gates + (size_t)b * G4;
    const float gi = g[j];
    const float gf = g[j + HID];
    const float gg = g[j + 2 * HID];
    const float go = g[j + 3 * HID];
    const float si = 1.0f / (1.0f + __expf(-gi));
    const float sf = 1.0f / (1.0f + __expf(-gf));
    const float so = 1.0f / (1.0f + __expf(-go));
    const float cn = sf * c[idx] + si * tanhf(gg);
    const float hn = so * tanhf(cn);
    c[idx] = cn;
    h[idx] = hn;
    hb[idx] = (__bf16)hn;
}

// ---------------- deterministic loss reduction ----------------
__global__ __launch_bounds__(256)
void loss_reduce(const float* __restrict__ picked, float* __restrict__ out)
{
    __shared__ float s[256];
    const int tid = threadIdx.x;
    float acc = 0.0f;
    for (int i = 0; i < 64; ++i) acc += picked[tid * 64 + i];   // 256*64 == 16384
    s[tid] = acc;
    __syncthreads();
    for (int ss = 128; ss > 0; ss >>= 1) { if (tid < ss) s[tid] += s[tid + ss]; __syncthreads(); }
    if (tid == 0) out[0] = -s[0] / (float)(MSG_LEN * BATCH);
}

// ---------------- host orchestration ----------------
extern "C" void kernel_launch(void* const* d_in, const int* in_sizes, int n_in,
                              void* d_out, int out_size, void* d_ws, size_t ws_size,
                              hipStream_t stream)
{
    const float* targets = (const float*)d_in[0];
    const float* W_attr  = (const float*)d_in[1];
    const float* b_attr  = (const float*)d_in[2];
    const float* W_ih    = (const float*)d_in[3];
    const float* W_hh    = (const float*)d_in[4];
    const float* b_ih    = (const float*)d_in[5];
    const float* b_hh    = (const float*)d_in[6];
    const float* W_hv    = (const float*)d_in[7];
    const float* b_hv    = (const float*)d_in[8];
    const int*   trg     = (const int*)d_in[9];

    char* ws = (char*)d_ws;
    size_t off = 0;
    auto carve = [&](size_t bytes) -> char* {
        char* p = ws + off;
        off += (bytes + 255) & ~(size_t)255;
        return p;
    };
    __bf16* tgt_bf   = (__bf16*)carve((size_t)BATCH * ATTR * 2);
    __bf16* Wattr_bf = (__bf16*)carve((size_t)HID * ATTR * 2);
    __bf16* Whh_bf   = (__bf16*)carve((size_t)G4 * HID * 2);
    __bf16* Whv_bf   = (__bf16*)carve((size_t)VOCAB * HID * 2);
    float*  h        = (float*) carve((size_t)BATCH * HID * 4);
    __bf16* h_bf     = (__bf16*)carve((size_t)BATCH * HID * 2);
    float*  c        = (float*) carve((size_t)BATCH * HID * 4);
    float*  gates    = (float*) carve((size_t)BATCH * G4 * 4);
    float*  picked   = (float*) carve((size_t)MSG_LEN * BATCH * 4);
    int*    tokens   = (int*)   carve((size_t)BATCH * 4);

    float* out       = (float*)d_out;
    float* loss_out  = out;
    float* msg_out   = out + 1;
    float* logp_base = out + 1 + (size_t)BATCH * MSG_LEN;

    // bf16 weight / input conversion (one-time; graph replays it deterministically)
    f32_to_bf16_k<<<(BATCH * ATTR + 255) / 256, 256, 0, stream>>>(targets, tgt_bf, BATCH * ATTR);
    f32_to_bf16_k<<<(HID * ATTR + 255) / 256, 256, 0, stream>>>(W_attr, Wattr_bf, HID * ATTR);
    f32_to_bf16_k<<<(G4 * HID + 255) / 256, 256, 0, stream>>>(W_hh, Whh_bf, G4 * HID);
    f32_to_bf16_k<<<(VOCAB * HID + 255) / 256, 256, 0, stream>>>(W_hv, Whv_bf, VOCAB * HID);
    fill_zero_f32_k<<<(BATCH * HID) / 256, 256, 0, stream>>>(c, BATCH * HID);

    // h0 = targets @ W_attr^T + b_attr   (M=1024, N=512, K=256)
    {
        int waves = (BATCH / 32) * (HID / 64);
        wmma_gemm_bf16<<<waves / 8, 256, 0, stream>>>(
            tgt_bf, Wattr_bf, b_attr, nullptr, nullptr, nullptr,
            h, h_bf, BATCH, HID, ATTR);
    }
    // first cell: x = 0 -> gates = h0 @ W_hh^T + b_ih + b_hh   (M=1024, N=2048, K=512)
    {
        int waves = (BATCH / 32) * (G4 / 64);
        wmma_gemm_bf16<<<waves / 8, 256, 0, stream>>>(
            h_bf, Whh_bf, b_ih, b_hh, nullptr, nullptr,
            gates, nullptr, BATCH, G4, HID);
    }
    lstm_update<<<(BATCH * HID) / 256, 256, 0, stream>>>(gates, c, h, h_bf);

    for (int t = 0; t < MSG_LEN; ++t) {
        float* logits = logp_base + (size_t)t * BATCH * VOCAB;
        // logits = h @ W_hv^T + b_hv   (M=1024, N=8192, K=512)
        int waves = (BATCH / 32) * (VOCAB / 64);
        wmma_gemm_bf16<<<waves / 8, 256, 0, stream>>>(
            h_bf, Whv_bf, b_hv, nullptr, nullptr, nullptr,
            logits, nullptr, BATCH, VOCAB, HID);

        logsoftmax_argmax<<<BATCH, 256, 0, stream>>>(logits, trg, t, msg_out, picked, tokens);

        if (t < MSG_LEN - 1) {   // reference discards the final step's h2/c2
            // gates = W_ih[:,tok] + h @ W_hh^T + b_ih + b_hh
            int w2 = (BATCH / 32) * (G4 / 64);
            wmma_gemm_bf16<<<w2 / 8, 256, 0, stream>>>(
                h_bf, Whh_bf, b_ih, b_hh, W_ih, tokens,
                gates, nullptr, BATCH, G4, HID);
            lstm_update<<<(BATCH * HID) / 256, 256, 0, stream>>>(gates, c, h, h_bf);
        }
    }

    loss_reduce<<<1, 256, 0, stream>>>(picked, loss_out);
}
